// LatentAdapterBlock_46918222742112
// MI455X (gfx1250) — matrix operand order repaired and checked by reading, not compiled, not run on hardware
//
#include <hip/hip_runtime.h>
#include <math.h>

typedef __bf16 bf16;
typedef __attribute__((ext_vector_type(16))) __bf16 bf16x16;
typedef __attribute__((ext_vector_type(8)))  __bf16 bf16x8;
typedef __attribute__((ext_vector_type(8)))  float  f32x8;

#define D_MODEL 2048
#define D_Z     512
#define N_HEADS 16
#define DH      128
#define B_SZ    4
#define T_SEQ   4096
#define M_LAT   64
#define DFF     8192
#define ROWS    (B_SZ * T_SEQ)   // 16384
#define ZROWS   (B_SZ * M_LAT)   // 256

// ---------------- WMMA fragment loaders (bf16, 16x16x32) ----------------
// A (16x32, row-major source): lane L holds row L&15; hi = L>>4 selects the
// K-halves {8*hi..8*hi+7} and {16+8*hi..23+8*hi} -> two contiguous 16B loads.
template <typename PT>
static __device__ __forceinline__ bf16x16 load_frag_a(PT tile, int ld) {
    const int l = threadIdx.x & 31;
    const int row = l & 15, hi = l >> 4;
    PT p = tile + (size_t)row * ld + hi * 8;
    union { bf16x16 v; bf16x8 h[2]; } u;
    u.h[0] = *(const bf16x8*)(p);
    u.h[1] = *(const bf16x8*)(p + 16);
    return u.v;
}

// B (32x16) read from a K-major ("transposed", [N][K]) buffer: lane L holds
// column L&15, K = {8*hi..8*hi+7} and {16+8*hi..23+8*hi} -> two 16B loads.
template <typename PT>
static __device__ __forceinline__ bf16x16 load_frag_b(PT tileT, int ld) {
    const int l = threadIdx.x & 31;
    const int col = l & 15, kh = l >> 4;
    PT p = tileT + (size_t)col * ld + kh * 8;
    union { bf16x16 v; bf16x8 h[2]; } u;
    u.h[0] = *(const bf16x8*)(p);
    u.h[1] = *(const bf16x8*)(p + 16);
    return u.v;
}

static __device__ __forceinline__ f32x8 wmma_bf16(bf16x16 a, bf16x16 b, f32x8 c) {
    return __builtin_amdgcn_wmma_f32_16x16x32_bf16(false, a, false, b,
                                                   (short)0, c, false, false);
}

// Async global->LDS copy of 16 bytes per lane (CDNA5 ASYNCcnt path).
static __device__ __forceinline__ void async_copy_b128(const bf16* gptr,
                                                       bf16* lptr) {
    const unsigned lds_off = (unsigned)(uintptr_t)lptr;  // low 32b = LDS addr
    asm volatile("global_load_async_to_lds_b128 %0, %1, off"
                 :: "v"(lds_off), "v"(gptr) : "memory");
}
static __device__ __forceinline__ void async_wait0() {
    asm volatile("s_wait_asynccnt 0x0" ::: "memory");
}

// ---------------- LayerNorm -> bf16 ----------------
__global__ __launch_bounds__(256)
void ln_bf16(const float* __restrict__ hp, const float* __restrict__ g,
             const float* __restrict__ be, bf16* __restrict__ out) {
    __shared__ float reds[8], redq[8];
    const int tid = threadIdx.x;
    const size_t row = blockIdx.x;
    const float* hr = hp + row * (size_t)D_MODEL;
    float4 v0 = ((const float4*)hr)[tid * 2];
    float4 v1 = ((const float4*)hr)[tid * 2 + 1];
    float s = v0.x + v0.y + v0.z + v0.w + v1.x + v1.y + v1.z + v1.w;
    float q = v0.x * v0.x + v0.y * v0.y + v0.z * v0.z + v0.w * v0.w +
              v1.x * v1.x + v1.y * v1.y + v1.z * v1.z + v1.w * v1.w;
    #pragma unroll
    for (int off = 16; off >= 1; off >>= 1) {
        s += __shfl_xor(s, off, 32);
        q += __shfl_xor(q, off, 32);
    }
    if ((tid & 31) == 0) { reds[tid >> 5] = s; redq[tid >> 5] = q; }
    __syncthreads();
    float ts = 0.f, tq = 0.f;
    #pragma unroll
    for (int i = 0; i < 8; ++i) { ts += reds[i]; tq += redq[i]; }
    const float mean = ts * (1.0f / D_MODEL);
    const float var  = tq * (1.0f / D_MODEL) - mean * mean;
    const float inv  = rsqrtf(var + 1e-5f);
    bf16* orow = out + row * (size_t)D_MODEL;
    const float* fv0 = (const float*)&v0;
    const float* fv1 = (const float*)&v1;
    #pragma unroll
    for (int j = 0; j < 8; ++j) {
        int c = tid * 8 + j;
        float x = (j < 4) ? fv0[j] : fv1[j - 4];
        orow[c] = (bf16)((x - mean) * inv * g[c] + be[c]);
    }
}

// ---------------- elementwise fp32 -> bf16 ----------------
__global__ __launch_bounds__(256)
void cvt_bf16(const float* __restrict__ in, bf16* __restrict__ out, size_t n) {
    size_t i = (size_t)blockIdx.x * 256 + threadIdx.x;
    if (i < n) out[i] = (bf16)in[i];
}

// ---------------- fp32 W[K][N] -> bf16 WT[N][K] ----------------
__global__ __launch_bounds__(256)
void transpose_bf16(const float* __restrict__ W, bf16* __restrict__ WT,
                    int Kd, int Nd) {
    size_t i = (size_t)blockIdx.x * 256 + threadIdx.x;
    if (i >= (size_t)Kd * Nd) return;
    int k = (int)(i / Nd), n = (int)(i % Nd);
    WT[(size_t)n * Kd + k] = (bf16)W[i];
}

// ---------------- generic WMMA GEMM: C = A[rows,K] * BT[N,K]^T + bias ------
// Block: 256 threads = 8 waves; block tile 64(M) x 256(N); wave tile 32x64
// (2 A-frags x 4 B-frags -> 8 WMMAs per 12 b128 loads, ~21 FLOP/B).
// EPI 0: bias -> bf16         EPI 1: bias + exact GELU -> bf16
// EPI 2: out = resid + tanh(alpha)*(x+bias) -> fp32
// EPI 3: bias -> bf16, stored transposed per batch: VT[b][col][row%64]
template <int EPI>
__global__ __launch_bounds__(256)
void gemm_bf16_wmma(const bf16* __restrict__ A, const bf16* __restrict__ BT,
                    const float* __restrict__ bias, int K, int N,
                    bf16* __restrict__ outb, float* __restrict__ outf,
                    const float* __restrict__ resid,
                    const float* __restrict__ alpha) {
    const int lane = threadIdx.x & 31;
    const int wave = threadIdx.x >> 5;
    const int wm = wave >> 2, wn = wave & 3;
    const size_t row0 = (size_t)blockIdx.y * 64 + wm * 32;
    const int    col0 = blockIdx.x * 256 + wn * 64;

    f32x8 acc[2][4] = {};
    const bf16* arow0 = A + row0 * K;
    const bf16* arow1 = arow0 + (size_t)16 * K;
    #pragma unroll 2
    for (int k0 = 0; k0 < K; k0 += 32) {
        bf16x16 a0 = load_frag_a(arow0 + k0, K);
        bf16x16 a1 = load_frag_a(arow1 + k0, K);
        #pragma unroll
        for (int nt = 0; nt < 4; ++nt) {
            bf16x16 b = load_frag_b(BT + (size_t)(col0 + nt * 16) * K + k0, K);
            acc[0][nt] = wmma_bf16(a0, b, acc[0][nt]);
            acc[1][nt] = wmma_bf16(a1, b, acc[1][nt]);
        }
    }
    const int hi = lane >> 4, cl = lane & 15;
    float sres = 0.f;
    if (EPI == 2) sres = tanhf(alpha[0]);
    #pragma unroll
    for (int mt = 0; mt < 2; ++mt) {
        #pragma unroll
        for (int nt = 0; nt < 4; ++nt) {
            const int col = col0 + nt * 16 + cl;
            const float bv = bias[col];
            #pragma unroll
            for (int r = 0; r < 8; ++r) {
                const size_t row = row0 + mt * 16 + r + 8 * hi;
                float v = acc[mt][nt][r] + bv;
                if (EPI == 0) {
                    outb[row * N + col] = (bf16)v;
                } else if (EPI == 1) {
                    v = 0.5f * v * (1.0f + erff(v * 0.70710678118654752f));
                    outb[row * N + col] = (bf16)v;
                } else if (EPI == 2) {
                    const size_t i = row * N + col;
                    outf[i] = resid[i] + sres * v;
                } else {  // EPI == 3: V stored transposed: [b][d][m]
                    const int bb = (int)(row >> 6), m = (int)(row & 63);
                    outb[((size_t)bb * N + col) * M_LAT + m] = (bf16)v;
                }
            }
        }
    }
}

// ---------------- fused cross-attention (M=64 keys, dh=128) ----------------
// One wave per 16-row query tile; 8 tiles per block all share (batch, head)
// since 256 tiles/head % 8 == 0. K and V^T tiles are staged into LDS once per
// block via global_load_async_to_lds_b128 (ASYNCcnt), then all B-fragments
// come from bank-conflict-free LDS (pitches 136 / 72 elements).
__global__ __launch_bounds__(256)
void attn_wmma(const bf16* __restrict__ Q, const bf16* __restrict__ Kk,
               const bf16* __restrict__ VT, bf16* __restrict__ ctx) {
    __shared__ __attribute__((aligned(16))) bf16 sK[M_LAT][136];   // K tile
    __shared__ __attribute__((aligned(16))) bf16 sV[DH][72];       // V^T tile
    __shared__ __attribute__((aligned(16))) bf16 sbuf[8][16][72];  // softmax w
    const int tid  = threadIdx.x;
    const int lane = tid & 31;
    const int wave = tid >> 5;
    const int tg0 = blockIdx.x * 8;
    const int b   = tg0 >> 12;                 // tiles per batch = 16*256
    const int hh  = (tg0 & 4095) >> 8;
    const int t0  = (((tg0 & 255) + wave)) << 4;
    const int hi = lane >> 4, cl = lane & 15;

    // ---- stage K (64x128) and V^T (128x64) into LDS, 16B per lane ----
    #pragma unroll
    for (int it = 0; it < 4; ++it) {
        const int chunk = it * 256 + tid;      // 0..1023
        const int m  = chunk >> 4;             // 16 chunks per 128-wide row
        const int d8 = (chunk & 15) * 8;
        async_copy_b128(Kk + ((size_t)b * M_LAT + m) * D_MODEL + hh * DH + d8,
                        &sK[m][d8]);
    }
    #pragma unroll
    for (int it = 0; it < 4; ++it) {
        const int chunk = it * 256 + tid;
        const int d  = chunk >> 3;             // 8 chunks per 64-wide row
        const int m8 = (chunk & 7) * 8;
        async_copy_b128(VT + ((size_t)b * D_MODEL + hh * DH + d) * M_LAT + m8,
                        &sV[d][m8]);
    }
    async_wait0();
    __syncthreads();

    // ---- scores S[16,64] = (Q tile)(K tile)^T, dh=128 in 4 chunks of 32 ----
    const bf16* qbase = Q + ((size_t)b * T_SEQ + t0) * D_MODEL + hh * DH;
    bf16x16 afr[4];
    #pragma unroll
    for (int kk = 0; kk < 4; ++kk) afr[kk] = load_frag_a(qbase + kk * 32, D_MODEL);
    f32x8 s[4] = {};
    #pragma unroll
    for (int nt = 0; nt < 4; ++nt) {
        #pragma unroll
        for (int kk = 0; kk < 4; ++kk) {
            bf16x16 bfr = load_frag_b(&sK[nt * 16][kk * 32], 136);
            s[nt] = wmma_bf16(afr[kk], bfr, s[nt]);
        }
    }

    // ---- softmax over 64 cols; rows r+8*hi live in 16-lane halves ----
    const float scale = 0.08838834764831845f;  // 1/sqrt(128)
    #pragma unroll
    for (int r = 0; r < 8; ++r) {
        float m = fmaxf(fmaxf(s[0][r], s[1][r]), fmaxf(s[2][r], s[3][r]));
        #pragma unroll
        for (int off = 1; off < 16; off <<= 1) m = fmaxf(m, __shfl_xor(m, off, 32));
        float sum = 0.f;
        #pragma unroll
        for (int nt = 0; nt < 4; ++nt) {
            float e = __expf((s[nt][r] - m) * scale);
            s[nt][r] = e;
            sum += e;
        }
        #pragma unroll
        for (int off = 1; off < 16; off <<= 1) sum += __shfl_xor(sum, off, 32);
        const float rinv = 1.0f / sum;
        #pragma unroll
        for (int nt = 0; nt < 4; ++nt)
            sbuf[wave][r + 8 * hi][nt * 16 + cl] = (bf16)(s[nt][r] * rinv);
    }
    __syncthreads();

    // ---- ctx[16,128] = W[16,64] @ V[64,128] (V from LDS, [d][m] layout) ----
    f32x8 c[8] = {};
    #pragma unroll
    for (int kk = 0; kk < 2; ++kk) {
        bf16x16 a2 = load_frag_a(&sbuf[wave][0][kk * 32], 72);
        #pragma unroll
        for (int dn = 0; dn < 8; ++dn) {
            bf16x16 b2 = load_frag_b(&sV[dn * 16][kk * 32], 72);
            c[dn] = wmma_bf16(a2, b2, c[dn]);
        }
    }
    #pragma unroll
    for (int dn = 0; dn < 8; ++dn) {
        const int col = hh * DH + dn * 16 + cl;
        #pragma unroll
        for (int r = 0; r < 8; ++r) {
            const size_t row = (size_t)b * T_SEQ + t0 + r + 8 * hi;
            ctx[row * D_MODEL + col] = (bf16)c[dn][r];
        }
    }
}

// ---------------- launcher ----------------
extern "C" void kernel_launch(void* const* d_in, const int* in_sizes, int n_in,
                              void* d_out, int out_size, void* d_ws, size_t ws_size,
                              hipStream_t stream) {
    const float* h     = (const float*)d_in[0];
    const float* z     = (const float*)d_in[1];
    const float* ln_g  = (const float*)d_in[2];
    const float* ln_b  = (const float*)d_in[3];
    const float* Wq    = (const float*)d_in[4];
    const float* bq    = (const float*)d_in[5];
    const float* Wk    = (const float*)d_in[6];
    const float* bk    = (const float*)d_in[7];
    const float* Wv    = (const float*)d_in[8];
    const float* bv    = (const float*)d_in[9];
    const float* W1    = (const float*)d_in[10];
    const float* b1    = (const float*)d_in[11];
    const float* W2    = (const float*)d_in[12];
    const float* b2    = (const float*)d_in[13];
    const float* alpha = (const float*)d_in[14];
    float* out = (float*)d_out;

    char* ws = (char*)d_ws;
    size_t off = 0;
    auto alloc = [&](size_t bytes) -> char* {
        char* p = ws + off;
        off += (bytes + 255) & ~(size_t)255;
        return p;
    };
    bf16* hq  = (bf16*)alloc((size_t)ROWS * D_MODEL * 2);
    bf16* Qb  = (bf16*)alloc((size_t)ROWS * D_MODEL * 2);
    bf16* ctx = (bf16*)alloc((size_t)ROWS * D_MODEL * 2);
    bf16* ffn = (bf16*)alloc((size_t)ROWS * DFF * 2);
    bf16* zb  = (bf16*)alloc((size_t)ZROWS * D_Z * 2);
    bf16* Kb  = (bf16*)alloc((size_t)ZROWS * D_MODEL * 2);
    bf16* VTt = (bf16*)alloc((size_t)B_SZ * D_MODEL * M_LAT * 2);
    bf16* WqT = (bf16*)alloc((size_t)D_MODEL * D_MODEL * 2);
    bf16* WkT = (bf16*)alloc((size_t)D_MODEL * D_Z * 2);
    bf16* WvT = (bf16*)alloc((size_t)D_MODEL * D_Z * 2);
    bf16* W1T = (bf16*)alloc((size_t)DFF * D_MODEL * 2);
    bf16* W2T = (bf16*)alloc((size_t)D_MODEL * DFF * 2);

    // Stage 0: precision prep
    ln_bf16<<<dim3(ROWS), dim3(256), 0, stream>>>(h, ln_g, ln_b, hq);
    const size_t nz = (size_t)ZROWS * D_Z;
    cvt_bf16<<<dim3((unsigned)((nz + 255) / 256)), dim3(256), 0, stream>>>(z, zb, nz);
    transpose_bf16<<<dim3((unsigned)(((size_t)D_MODEL * D_MODEL + 255) / 256)), dim3(256), 0, stream>>>(Wq, WqT, D_MODEL, D_MODEL);
    transpose_bf16<<<dim3((unsigned)(((size_t)D_Z * D_MODEL + 255) / 256)), dim3(256), 0, stream>>>(Wk, WkT, D_Z, D_MODEL);
    transpose_bf16<<<dim3((unsigned)(((size_t)D_Z * D_MODEL + 255) / 256)), dim3(256), 0, stream>>>(Wv, WvT, D_Z, D_MODEL);
    transpose_bf16<<<dim3((unsigned)(((size_t)D_MODEL * DFF + 255) / 256)), dim3(256), 0, stream>>>(W1, W1T, D_MODEL, DFF);
    transpose_bf16<<<dim3((unsigned)(((size_t)DFF * D_MODEL + 255) / 256)), dim3(256), 0, stream>>>(W2, W2T, DFF, D_MODEL);

    // Stage 1: projections
    gemm_bf16_wmma<0><<<dim3(D_MODEL / 256, ROWS / 64), dim3(256), 0, stream>>>(
        hq, WqT, bq, D_MODEL, D_MODEL, Qb, nullptr, nullptr, nullptr);
    gemm_bf16_wmma<0><<<dim3(D_MODEL / 256, ZROWS / 64), dim3(256), 0, stream>>>(
        zb, WkT, bk, D_Z, D_MODEL, Kb, nullptr, nullptr, nullptr);
    gemm_bf16_wmma<3><<<dim3(D_MODEL / 256, ZROWS / 64), dim3(256), 0, stream>>>(
        zb, WvT, bv, D_Z, D_MODEL, VTt, nullptr, nullptr, nullptr);

    // Stage 2: cross-attention (one wave per 16 query rows)
    attn_wmma<<<dim3(B_SZ * N_HEADS * (T_SEQ / 16) / 8), dim3(256), 0, stream>>>(
        Qb, Kb, VTt, ctx);

    // Stage 3: FFN
    gemm_bf16_wmma<1><<<dim3(DFF / 256, ROWS / 64), dim3(256), 0, stream>>>(
        ctx, W1T, b1, D_MODEL, DFF, ffn, nullptr, nullptr, nullptr);
    gemm_bf16_wmma<2><<<dim3(D_MODEL / 256, ROWS / 64), dim3(256), 0, stream>>>(
        ffn, W2T, b2, DFF, D_MODEL, nullptr, out, h, alpha);
}
